// GraphAttentionLayer_90872918049323
// MI455X (gfx1250) — matrix operand order repaired and checked by reading, not compiled, not run on hardware
//
#include <hip/hip_runtime.h>

// GraphAttention for MI455X (gfx1250, wave32, WMMA bf16 16x16x32).
// Flash-style: kernel1 projects x@W -> h (bf16, row-major + transposed copy),
// kernel2 computes per-row softmax stats (max, 1/sum) with WMMA score tiles,
// kernel3 recomputes score tiles, forms P in-register (C-layout of the score
// WMMA == A-layout of P^T for the output WMMA, no shuffles), and accumulates
// out_j = sum_i P[i,j] * h[i] with two more WMMAs per 32-i chunk.

#define S_DIM 4096
#define F_DIM 128
#define H_DIM 4
#define K_DIM 32
#define B_DIM 2
#define BH_DIM (B_DIM * H_DIM)
#define ALPHA 0.2f
#define INV_SQRT_K 0.17677669529663688f
#define LOG2E 1.4426950408889634f

typedef __attribute__((ext_vector_type(16))) __bf16 v16bf;
typedef __attribute__((ext_vector_type(8)))  float  v8f;
typedef __attribute__((ext_vector_type(4)))  unsigned int u32x4;
typedef __attribute__((ext_vector_type(4)))  float  f32x4;

union BV { v16bf v; u32x4 q[2]; unsigned short u[16]; };

static __device__ __forceinline__ unsigned short f2bf(float f) {
  unsigned int u = __float_as_uint(f);
  return (unsigned short)((u + 0x7fffu + ((u >> 16) & 1u)) >> 16);
}

static __device__ __forceinline__ v8f wmma_bf16(v16bf a, v16bf b, v8f c) {
  // D = A(16x32 bf16) x B(32x16 bf16) + C(16x16 f32)
  return __builtin_amdgcn_wmma_f32_16x16x32_bf16(false, a, false, b,
                                                 (short)0, c, false, false);
}

// ---------------------------------------------------------------------------
// Kernel 1: h[b,s,h,k] = sum_f x[b,s,f] * W[h,f,k]   (fp32 dot, bf16 store)
// One wave per (b,s,h); lane = k (32 lanes == K_DIM). Coalesced W reads.
// Writes hb[bh][s][k] (row-major) and hbT[bh][k][s] (transposed).
// ---------------------------------------------------------------------------
__global__ __launch_bounds__(256)
void proj_kernel(const float* __restrict__ x, const float* __restrict__ W,
                 unsigned short* __restrict__ hb, unsigned short* __restrict__ hbT) {
  int wid  = (blockIdx.x * blockDim.x + threadIdx.x) >> 5;
  int lane = threadIdx.x & 31;
  int h  = wid % H_DIM;
  int bs = wid / H_DIM;                 // b*S + s
  const float* xr = x + (long)bs * F_DIM;
  const float* wp = W + (long)h * F_DIM * K_DIM + lane; // W[h][f][lane]
  float acc = 0.f;
#pragma unroll 8
  for (int f = 0; f < F_DIM; ++f)
    acc = fmaf(xr[f], wp[(long)f * K_DIM], acc);
  int b = bs / S_DIM, s = bs % S_DIM;
  int bh = b * H_DIM + h;
  unsigned short v = f2bf(acc);
  hb [((long)bh * S_DIM + s) * K_DIM + lane] = v;
  hbT[((long)bh * K_DIM + lane) * S_DIM + s] = v;
}

// ---------------------------------------------------------------------------
// Kernel 2: per-row softmax stats over j. One wave per (bh, i-block of 16).
// Pass A: row max of leakyrelu(score); pass B: sum of exp2((s-m)*log2e).
// Score tile (16i x 16j) = ONE v_wmma_f32_16x16x32_bf16 (K = head dim = 32).
// ---------------------------------------------------------------------------
__global__ __launch_bounds__(256)
void stats_kernel(const unsigned short* __restrict__ hb,
                  float* __restrict__ mo, float* __restrict__ rlo) {
  int wid  = (blockIdx.x * blockDim.x + threadIdx.x) >> 5;
  int lane = threadIdx.x & 31;
  int l16 = lane & 15, hi = lane >> 4;
  int bh = wid >> 8;                       // / (S/16)
  int i0 = (wid & 255) << 4;
  const unsigned short* base = hb + (long)bh * S_DIM * K_DIM;

  // A tile: rows i0+l16; lanes<16 hold K {0..7,16..23}, lanes>=16 {8..15,24..31}
  BV a;
  const unsigned short* ar = base + (long)(i0 + l16) * K_DIM + 8 * hi;
  a.q[0] = *(const u32x4*)(ar);
  a.q[1] = *(const u32x4*)(ar + 16);

  float m[8];
#pragma unroll
  for (int r = 0; r < 8; ++r) m[r] = -3.0e38f;

  for (int j0 = 0; j0 < S_DIM; j0 += 16) {
    BV bt;  // B tile = h[j]^T: lane holds col j0+l16, K-half selected by hi
    const unsigned short* br = base + (long)(j0 + l16) * K_DIM + 8 * hi;
    bt.q[0] = *(const u32x4*)(br);
    bt.q[1] = *(const u32x4*)(br + 16);
    v8f c = {};
    c = wmma_bf16(a.v, bt.v, c);
#pragma unroll
    for (int r = 0; r < 8; ++r) {
      float t = c[r] * INV_SQRT_K;
      t = fmaxf(t, ALPHA * t);             // LeakyReLU
      m[r] = fmaxf(m[r], t);
    }
  }
  // reduce max across the 16 lanes that share each row (xor<16 stays in half)
#pragma unroll
  for (int off = 1; off < 16; off <<= 1) {
#pragma unroll
    for (int r = 0; r < 8; ++r) m[r] = fmaxf(m[r], __shfl_xor(m[r], off, 32));
  }

  float l[8];
#pragma unroll
  for (int r = 0; r < 8; ++r) l[r] = 0.f;
  for (int j0 = 0; j0 < S_DIM; j0 += 16) {
    BV bt;
    const unsigned short* br = base + (long)(j0 + l16) * K_DIM + 8 * hi;
    bt.q[0] = *(const u32x4*)(br);
    bt.q[1] = *(const u32x4*)(br + 16);
    v8f c = {};
    c = wmma_bf16(a.v, bt.v, c);
#pragma unroll
    for (int r = 0; r < 8; ++r) {
      float t = c[r] * INV_SQRT_K;
      t = fmaxf(t, ALPHA * t);
      l[r] += exp2f((t - m[r]) * LOG2E);
    }
  }
#pragma unroll
  for (int off = 1; off < 16; off <<= 1) {
#pragma unroll
    for (int r = 0; r < 8; ++r) l[r] += __shfl_xor(l[r], off, 32);
  }

  if (l16 == 0) {  // lanes 0 and 16 each own 8 consecutive rows
    float* mp = mo  + (long)bh * S_DIM + i0 + 8 * hi;
    float* rp = rlo + (long)bh * S_DIM + i0 + 8 * hi;
    f32x4 m0 = {m[0], m[1], m[2], m[3]}, m1 = {m[4], m[5], m[6], m[7]};
    f32x4 r0 = {1.f/l[0], 1.f/l[1], 1.f/l[2], 1.f/l[3]};
    f32x4 r1 = {1.f/l[4], 1.f/l[5], 1.f/l[6], 1.f/l[7]};
    *(f32x4*)(mp) = m0;  *(f32x4*)(mp + 4) = m1;
    *(f32x4*)(rp) = r0;  *(f32x4*)(rp + 4) = r1;
  }
}

// ---------------------------------------------------------------------------
// Kernel 3: out[b,j,h,:] = sum_i P[i,j] * h[b,i,h,:]. One wave per (bh, j16).
// Per 32-i chunk: 2 score WMMAs -> P (exp2 * 1/l) packed in-register as the
// A-operand of P^T (layout identity, no shuffles) -> 2 output WMMAs using
// B = h-chunk loaded contiguously from hbT.
// ---------------------------------------------------------------------------
__global__ __launch_bounds__(256)
void out_kernel(const unsigned short* __restrict__ hb,
                const unsigned short* __restrict__ hbT,
                const float* __restrict__ m_arr, const float* __restrict__ rl_arr,
                float* __restrict__ out) {
  int wid  = (blockIdx.x * blockDim.x + threadIdx.x) >> 5;
  int lane = threadIdx.x & 31;
  int l16 = lane & 15, hi = lane >> 4;
  int bh = wid >> 8;
  int j0 = (wid & 255) << 4;
  int b = bh / H_DIM, h = bh % H_DIM;

  const unsigned short* base  = hb  + (long)bh * S_DIM * K_DIM;
  const unsigned short* tbase = hbT + (long)bh * K_DIM * S_DIM;
  const float* mrow = m_arr  + (long)bh * S_DIM;
  const float* rrow = rl_arr + (long)bh * S_DIM;

  // Score B tile for this j-block (reused over all i): h[j]^T
  BV bj;
  const unsigned short* br = base + (long)(j0 + l16) * K_DIM + 8 * hi;
  bj.q[0] = *(const u32x4*)(br);
  bj.q[1] = *(const u32x4*)(br + 16);

  v8f acc1 = {}; v8f acc2 = {};  // out tile: rows j, cols k 0..15 / 16..31

  for (int i0 = 0; i0 < S_DIM; i0 += 32) {
    // score tiles for i0..i0+15 and i0+16..i0+31
    BV a1, a2;
    const unsigned short* ar1 = base + (long)(i0 + l16) * K_DIM + 8 * hi;
    a1.q[0] = *(const u32x4*)(ar1);
    a1.q[1] = *(const u32x4*)(ar1 + 16);
    const unsigned short* ar2 = base + (long)(i0 + 16 + l16) * K_DIM + 8 * hi;
    a2.q[0] = *(const u32x4*)(ar2);
    a2.q[1] = *(const u32x4*)(ar2 + 16);
    v8f z = {};
    v8f c1 = wmma_bf16(a1.v, bj.v, z);
    v8f c2 = wmma_bf16(a2.v, bj.v, z);

    // row stats: tile1 rows i0+8*hi+r, tile2 rows i0+16+8*hi+r
    float m1[8], r1[8], m2[8], r2[8];
    *(f32x4*)&m1[0] = *(const f32x4*)(mrow + i0 + 8 * hi);
    *(f32x4*)&m1[4] = *(const f32x4*)(mrow + i0 + 8 * hi + 4);
    *(f32x4*)&r1[0] = *(const f32x4*)(rrow + i0 + 8 * hi);
    *(f32x4*)&r1[4] = *(const f32x4*)(rrow + i0 + 8 * hi + 4);
    *(f32x4*)&m2[0] = *(const f32x4*)(mrow + i0 + 16 + 8 * hi);
    *(f32x4*)&m2[4] = *(const f32x4*)(mrow + i0 + 16 + 8 * hi + 4);
    *(f32x4*)&r2[0] = *(const f32x4*)(rrow + i0 + 16 + 8 * hi);
    *(f32x4*)&r2[4] = *(const f32x4*)(rrow + i0 + 16 + 8 * hi + 4);

    // P^T A-operand: elements 0..7 from tile1, 8..15 from tile2 (layout match)
    BV ap;
#pragma unroll
    for (int r = 0; r < 8; ++r) {
      float t = c1[r] * INV_SQRT_K; t = fmaxf(t, ALPHA * t);
      ap.u[r]     = f2bf(exp2f((t - m1[r]) * LOG2E) * r1[r]);
      float u = c2[r] * INV_SQRT_K; u = fmaxf(u, ALPHA * u);
      ap.u[8 + r] = f2bf(exp2f((u - m2[r]) * LOG2E) * r2[r]);
    }

    // Output B tiles: B[i_local][k] from hbT (contiguous in i)
    BV bo1, bo2;
    const unsigned short* t1 = tbase + (long)(l16) * S_DIM + i0 + 8 * hi;
    bo1.q[0] = *(const u32x4*)(t1);
    bo1.q[1] = *(const u32x4*)(t1 + 16);
    const unsigned short* t2 = tbase + (long)(16 + l16) * S_DIM + i0 + 8 * hi;
    bo2.q[0] = *(const u32x4*)(t2);
    bo2.q[1] = *(const u32x4*)(t2 + 16);

    acc1 = wmma_bf16(ap.v, bo1.v, acc1);
    acc2 = wmma_bf16(ap.v, bo2.v, acc2);
  }

  // store: acc vgpr r -> row j = j0 + r + 8*hi, col = h*32 + (0|16) + l16
#pragma unroll
  for (int r = 0; r < 8; ++r) {
    long j = j0 + r + 8 * hi;
    float* op = out + ((long)b * S_DIM + j) * (H_DIM * K_DIM) + h * K_DIM;
    op[l16]      = acc1[r];
    op[16 + l16] = acc2[r];
  }
}

extern "C" void kernel_launch(void* const* d_in, const int* in_sizes, int n_in,
                              void* d_out, int out_size, void* d_ws, size_t ws_size,
                              hipStream_t stream) {
  const float* x = (const float*)d_in[0];
  const float* W = (const float*)d_in[1];
  float* out = (float*)d_out;
  char* ws = (char*)d_ws;
  // ws layout: hb (2MB) | hbT (2MB) | m (128KB) | rl (128KB)
  unsigned short* hb  = (unsigned short*)(ws);
  unsigned short* hbT = (unsigned short*)(ws + (size_t)(2u << 20));
  float* m_arr  = (float*)(ws + (size_t)(4u << 20));
  float* rl_arr = (float*)(ws + (size_t)(4u << 20) + (size_t)(128u << 10));

  // kernel1: B*S*H = 32768 waves, 8 waves/block
  proj_kernel<<<4096, 256, 0, stream>>>(x, W, hb, hbT);
  // kernel2/3: BH * (S/16) = 2048 waves, 8 waves/block
  stats_kernel<<<256, 256, 0, stream>>>(hb, m_arr, rl_arr);
  out_kernel<<<256, 256, 0, stream>>>(hb, hbT, m_arr, rl_arr, out);
}
